// APJFNN_39968965656699
// MI455X (gfx1250) — matrix operand order repaired and checked by reading, not compiled, not run on hardware
//
#include <hip/hip_runtime.h>
#include <hip/hip_bf16.h>
#include <math.h>

// ---------------------------------------------------------------------------
// APJFNN forward for MI455X (gfx1250, wave32, WMMA f32_16x16x32_f16).
//
// Assumed flat input order (jax tree_flatten: dict keys sorted at each level):
//  0 geek_sent(32,12,50)i  1 job_sent(32,12,50)i
//  params:
//  2 co.U(256,256) 3 co.W(256,256) 4 co.v(256) 5 emb(30000,128)
//  geek_layer_lstm: 6 Whh_b 7 Whh_f 8 Wih_b 9 Wih_f 10 bhh_b 11 bhh_f 12 bih_b 13 bih_f
//  geek_lstm:       14 Whh_b 15 Whh_f 16 Wih_b 17 Wih_f 18 bhh_b 19 bhh_f 20 bih_b 21 bih_f
//  geek_self_attn:  22 W1 23 b1 24 v
//  job_attn1:       25 W1 26 b1 27 v
//  job_attn2:       28 W1 29 b1 30 v
//  job_layer_lstm:  31 Whh_b 32 Whh_f 33 Wih_b 34 Wih_f 35 bhh_b 36 bhh_f 37 bih_b 38 bih_f
//  job_lstm:        39 Whh_b 40 Whh_f 41 Wih_b 42 Wih_f 43 bhh_b 44 bhh_f 45 bih_b 46 bih_f
//  mlp:             47 W1(768,768) 48 W2(768,1) 49 b1(768) 50 b2(1)
// Output: (32,) float32.
// ---------------------------------------------------------------------------

typedef __attribute__((ext_vector_type(16))) _Float16 v16h;
typedef __attribute__((ext_vector_type(8)))  float    v8f;

union V16H { v16h h; uint4 u[2]; };

// Scheduler guidance (best-effort): group loads ahead of the WMMA chain so
// waits become partial (s_wait_loadcnt N) instead of full drains.
#if __has_builtin(__builtin_amdgcn_sched_group_barrier)
#define SGB(mask, n) __builtin_amdgcn_sched_group_barrier((mask), (n), 0)
#else
#define SGB(mask, n)
#endif
// masks: 0x002 VALU, 0x008 WMMA, 0x020 VMEM-read, 0x100 DS-read

__device__ __forceinline__ v8f wmma16(v16h a, v16h b, v8f c) {
  // D = A(16x32 f16) * B(32x16 f16) + C(16x16 f32)
  return __builtin_amdgcn_wmma_f32_16x16x32_f16(
      false, a, false, b, (short)0, c, false, false);
}

__device__ __forceinline__ float sigmoidf_(float x) {
  return 1.0f / (1.0f + __expf(-x));
}

// ------------------------- embedding gather (f32 -> f16 A matrix) ----------
__global__ void k_gather_emb(const int* __restrict__ tok,
                             const float* __restrict__ emb,
                             _Float16* __restrict__ out, int total) {
  int idx = blockIdx.x * 256 + threadIdx.x;
  if (idx >= total) return;
  int i = idx >> 7, d = idx & 127;
  out[idx] = (_Float16)emb[(size_t)tok[i] * 128 + d];
}

// ------------------------- f32 -> f16 flat convert -------------------------
__global__ void k_cvt16(const float* __restrict__ s, _Float16* __restrict__ d,
                        int n) {
  int i = blockIdx.x * 256 + threadIdx.x;
  if (i < n) d[i] = (_Float16)s[i];
}

// ------------- transpose+convert weight: src[K][N] -> dst[N][Ktot] ---------
__global__ void k_wT16(const float* __restrict__ src, _Float16* __restrict__ dst,
                       int K, int N, int Ktot, int kOff) {
  int idx = blockIdx.x * 256 + threadIdx.x;
  if (idx >= K * N) return;
  int k = idx / N, n = idx % N;
  dst[(size_t)n * Ktot + kOff + k] = (_Float16)src[idx];
}

// ------------------------- generic WMMA GEMM (row-major) -------------------
// C[M][N] = A[M][K](f16) * Bt[N][K](f16)^T (+bias) (opt tanh), f32 out.
// block = 128 threads (4 waves); wave w -> M-tile blockIdx.x*4+w;
// blockIdx.y -> 64 columns (4 N-tiles per wave).
__global__ void k_gemm(const _Float16* __restrict__ A,
                       const _Float16* __restrict__ Bt,
                       const float* __restrict__ bias,
                       float* __restrict__ C,
                       int M, int N, int K, int doTanh) {
  int lane = threadIdx.x & 31;
  int wave = threadIdx.x >> 5;
  int mt = blockIdx.x * 4 + wave;
  if (mt * 16 >= M) return;
  int lm = lane & 15;
  int hi = lane >> 4;             // half-wave select
  int nb = blockIdx.y * 64;
  const _Float16* Arow = A + (size_t)(mt * 16 + lm) * K;
  const _Float16* Brow0 = Bt + (size_t)(nb + lm) * K + hi * 16;
  v8f acc[4] = {};
  for (int k0 = 0; k0 < K; k0 += 32) {
    V16H a, b[4];
    a.u[0] = *(const uint4*)(Arow + k0 + hi * 16);       // K = hi*16 .. +7
    a.u[1] = *(const uint4*)(Arow + k0 + hi * 16 + 8);   // K = hi*16+8 .. +15
#pragma unroll
    for (int nt = 0; nt < 4; ++nt) {
      const _Float16* bp = Brow0 + (size_t)(nt * 16) * K + k0;
      b[nt].u[0] = *(const uint4*)bp;
      b[nt].u[1] = *(const uint4*)(bp + 8);              // K = hi*16 .. +15
    }
#pragma unroll
    for (int nt = 0; nt < 4; ++nt) acc[nt] = wmma16(a.h, b[nt].h, acc[nt]);
    SGB(0x020, 10);  // 2 A + 8 B b128 reads first
    SGB(0x008, 4);   // then the 4-WMMA chain
  }
#pragma unroll
  for (int nt = 0; nt < 4; ++nt) {
#pragma unroll
    for (int r = 0; r < 8; ++r) {
      int m = mt * 16 + r + hi * 8;
      int n = nb + nt * 16 + lm;
      float v = acc[nt][r];
      if (bias) v += bias[n];
      if (doTanh) v = tanhf(v);
      C[(size_t)m * N + n] = v;
    }
  }
}

// ------------------------- fused BiLSTM scan -------------------------------
// gates = x_t @ Wih + h @ Whh + bih + bhh, all via WMMA from the transposed
// f16 weight Wt[n][k], k in [0,Kx)=Wih, [Kx,Kx+128)=Whh.
// Each block owns 16 batch rows; 4 waves x 128 gate columns each;
// h (f16) / c / gates live in LDS; time loop is local (recurrence is
// row-independent, so no cross-workgroup sync is ever needed).
__global__ void k_lstm_scan(const _Float16* __restrict__ x16, int Kx, int T,
                            const _Float16* __restrict__ Wt_f,
                            const _Float16* __restrict__ Wt_b,
                            const float* __restrict__ bih_f,
                            const float* __restrict__ bhh_f,
                            const float* __restrict__ bih_b,
                            const float* __restrict__ bhh_b,
                            float* __restrict__ out /* [rows][T][256] */) {
  __shared__ _Float16 h16_lds[16][128];
  __shared__ float c_lds[16][128];
  __shared__ float g_lds[16][512];
  const int dir = blockIdx.y;
  const _Float16* Wt = dir ? Wt_b : Wt_f;
  const float* b1 = dir ? bih_b : bih_f;
  const float* b2 = dir ? bhh_b : bhh_f;
  const int seq0 = blockIdx.x * 16;
  const int lane = threadIdx.x & 31;
  const int wave = threadIdx.x >> 5;
  const int lm = lane & 15;
  const int hi = lane >> 4;
  const int Ktot = Kx + 128;
  // base of this lane's B-fragment column (N = wave*128 + nt*16 + lm)
  const _Float16* Brow0 = Wt + (size_t)(wave * 128 + lm) * Ktot + hi * 16;

  for (int p = threadIdx.x; p < 16 * 128; p += 128) {
    h16_lds[p >> 7][p & 127] = (_Float16)0.0f;
    c_lds[p >> 7][p & 127] = 0.0f;
  }
  __syncthreads();

  for (int t = 0; t < T; ++t) {
    const int tt = dir ? (T - 1 - t) : t;
    v8f acc[8] = {};
    const _Float16* Arow = x16 + ((size_t)(seq0 + lm) * T + tt) * Kx;
    // prefetch next timestep's input row (global_prefetch_b8)
    if (t + 1 < T) {
      const int tn = dir ? (T - 2 - t) : (t + 1);
      __builtin_prefetch(x16 + ((size_t)(seq0 + lm) * T + tn) * Kx, 0, 3);
    }
    // ---- x_t @ Wih : batch all loads for the K-slice, then WMMA chain ----
    for (int k0 = 0; k0 < Kx; k0 += 32) {
      V16H a, b[8];
      a.u[0] = *(const uint4*)(Arow + k0 + hi * 16);
      a.u[1] = *(const uint4*)(Arow + k0 + hi * 16 + 8);
#pragma unroll
      for (int nt = 0; nt < 8; ++nt) {
        const _Float16* bp = Brow0 + (size_t)(nt * 16) * Ktot + k0;
        b[nt].u[0] = *(const uint4*)bp;
        b[nt].u[1] = *(const uint4*)(bp + 8);
      }
#pragma unroll
      for (int nt = 0; nt < 8; ++nt) acc[nt] = wmma16(a.h, b[nt].h, acc[nt]);
      SGB(0x020, 18);  // 2 A + 16 B b128 reads first
      SGB(0x008, 8);   // then the 8-WMMA chain
    }
    // ---- h @ Whh (A = f16 h from LDS, two contiguous b128 reads) ----
    for (int k0 = 0; k0 < 128; k0 += 32) {
      V16H a, b[8];
      a.u[0] = *(const uint4*)&h16_lds[lm][k0 + hi * 16];
      a.u[1] = *(const uint4*)&h16_lds[lm][k0 + hi * 16 + 8];
#pragma unroll
      for (int nt = 0; nt < 8; ++nt) {
        const _Float16* bp = Brow0 + (size_t)(nt * 16) * Ktot + Kx + k0;
        b[nt].u[0] = *(const uint4*)bp;
        b[nt].u[1] = *(const uint4*)(bp + 8);
      }
#pragma unroll
      for (int nt = 0; nt < 8; ++nt) acc[nt] = wmma16(a.h, b[nt].h, acc[nt]);
      SGB(0x100, 2);   // h fragments from LDS
      SGB(0x020, 16);  // 16 B b128 reads
      SGB(0x008, 8);   // then the 8-WMMA chain
    }
    // ---- spill gate tiles to LDS (C/D layout: M = r + hi*8, N = lane&15) --
#pragma unroll
    for (int nt = 0; nt < 8; ++nt)
#pragma unroll
      for (int r = 0; r < 8; ++r)
        g_lds[r + hi * 8][wave * 128 + nt * 16 + lm] = acc[nt][r];
    __syncthreads();
    // ---- activations: i,f,g,o splits; update c,h; emit h ----
    for (int p = threadIdx.x; p < 16 * 128; p += 128) {
      int m = p >> 7, j = p & 127;
      float iG = g_lds[m][j] + b1[j] + b2[j];
      float fG = g_lds[m][128 + j] + b1[128 + j] + b2[128 + j];
      float gG = g_lds[m][256 + j] + b1[256 + j] + b2[256 + j];
      float oG = g_lds[m][384 + j] + b1[384 + j] + b2[384 + j];
      float c = sigmoidf_(fG) * c_lds[m][j] + sigmoidf_(iG) * tanhf(gG);
      float h = sigmoidf_(oG) * tanhf(c);
      c_lds[m][j] = c;
      h16_lds[m][j] = (_Float16)h;
      out[((size_t)(seq0 + m) * T + tt) * 256 + dir * 128 + j] = h;
    }
    __syncthreads();
  }
}

// -------- attention score: s[r] = sum_d tanh(tmp[r][d]+b1[d]) * v[d] -------
__global__ void k_attn_score(const float* __restrict__ tmp,
                             const float* __restrict__ b1,
                             const float* __restrict__ v,
                             float* __restrict__ s, int R) {
  __shared__ float red[256];
  int r = blockIdx.x;
  int d = threadIdx.x;
  red[d] = tanhf(tmp[(size_t)r * 256 + d] + b1[d]) * v[d];
  __syncthreads();
  for (int o = 128; o > 0; o >>= 1) {
    if (d < o) red[d] += red[d + o];
    __syncthreads();
  }
  if (d == 0) s[r] = red[0];
}

// -------- softmax over T + weighted sum: out[g][d] = sum_t a[t]*x[g][t][d] -
__global__ void k_attn_pool(const float* __restrict__ s,
                            const float* __restrict__ x,
                            float* __restrict__ out, int T) {
  __shared__ float w[64];
  int g = blockIdx.x, d = threadIdx.x;
  if (d == 0) {
    float mx = -1e30f;
    for (int t = 0; t < T; ++t) mx = fmaxf(mx, s[(size_t)g * T + t]);
    float sum = 0.0f;
    for (int t = 0; t < T; ++t) {
      float e = __expf(s[(size_t)g * T + t] - mx);
      w[t] = e;
      sum += e;
    }
    float inv = 1.0f / sum;
    for (int t = 0; t < T; ++t) w[t] *= inv;
  }
  __syncthreads();
  float acc = 0.0f;
  for (int t = 0; t < T; ++t) acc += w[t] * x[((size_t)g * T + t) * 256 + d];
  out[(size_t)g * 256 + d] = acc;
}

// -------- co-attention: per (b,g) block computes sr[b,g,:] -----------------
__global__ void k_co_attn(const float* __restrict__ Wx,   // [384][50][256]
                          const float* __restrict__ Us,   // [384][256] (b,j)
                          const float* __restrict__ v,    // [256]
                          const float* __restrict__ gvo,  // [384][50][256]
                          float* __restrict__ sr) {       // [384][256]
  __shared__ float red[256];
  __shared__ float sc[50];
  __shared__ float wacc[50];
  int bg = blockIdx.x;
  int b = bg / 12;
  int d = threadIdx.x;
  if (d < 50) wacc[d] = 0.0f;
  __syncthreads();
  for (int j = 0; j < 12; ++j) {
    float usd = Us[((size_t)b * 12 + j) * 256 + d];
    for (int l = 0; l < 50; ++l) {
      red[d] = tanhf(Wx[((size_t)bg * 50 + l) * 256 + d] + usd) * v[d];
      __syncthreads();
      for (int o = 128; o > 0; o >>= 1) {
        if (d < o) red[d] += red[d + o];
        __syncthreads();
      }
      if (d == 0) sc[l] = red[0];
      __syncthreads();
    }
    if (d == 0) {
      float mx = -1e30f;
      for (int l = 0; l < 50; ++l) mx = fmaxf(mx, sc[l]);
      float sum = 0.0f;
      for (int l = 0; l < 50; ++l) {
        float e = __expf(sc[l] - mx);
        sc[l] = e;
        sum += e;
      }
      float inv = 1.0f / (12.0f * sum);  // includes / Sj
      for (int l = 0; l < 50; ++l) wacc[l] += sc[l] * inv;
    }
    __syncthreads();
  }
  float acc = 0.0f;
  for (int l = 0; l < 50; ++l)
    acc += wacc[l] * gvo[((size_t)bg * 50 + l) * 256 + d];
  sr[(size_t)bg * 256 + d] = acc;
}

// -------- x = [gj, gr, gj-gr] as f16 for MLP GEMM --------------------------
__global__ void k_concat(const float* __restrict__ gj,
                         const float* __restrict__ gr,
                         _Float16* __restrict__ x16) {
  int idx = blockIdx.x * 256 + threadIdx.x;
  if (idx >= 32 * 768) return;
  int b = idx / 768, k = idx % 768;
  float v;
  if (k < 256)
    v = gj[b * 256 + k];
  else if (k < 512)
    v = gr[b * 256 + (k - 256)];
  else
    v = gj[b * 256 + (k - 512)] - gr[b * 256 + (k - 512)];
  x16[idx] = (_Float16)v;
}

// -------- final: out[b] = h[b,:] . W2 + b2 ---------------------------------
__global__ void k_final(const float* __restrict__ h,
                        const float* __restrict__ W2,
                        const float* __restrict__ b2, float* __restrict__ out) {
  __shared__ float red[256];
  int b = blockIdx.x, t = threadIdx.x;
  float acc = 0.0f;
  for (int k = t; k < 768; k += 256) acc += h[(size_t)b * 768 + k] * W2[k];
  red[t] = acc;
  __syncthreads();
  for (int o = 128; o > 0; o >>= 1) {
    if (t < o) red[t] += red[t + o];
    __syncthreads();
  }
  if (t == 0) out[b] = red[0] + b2[0];
}

// ===========================================================================
extern "C" void kernel_launch(void* const* d_in, const int* in_sizes, int n_in,
                              void* d_out, int out_size, void* d_ws,
                              size_t ws_size, hipStream_t stream) {
  (void)in_sizes; (void)n_in; (void)out_size; (void)ws_size;
  const int NG = 32 * 12 * 50;  // 19200 tokens per side

  const int* geek_sent = (const int*)d_in[0];
  const int* job_sent  = (const int*)d_in[1];
  const float* coU = (const float*)d_in[2];
  const float* coW = (const float*)d_in[3];
  const float* coV = (const float*)d_in[4];
  const float* emb = (const float*)d_in[5];
  // lstm param helper indices: {Whh_b, Whh_f, Wih_b, Wih_f, bhh_b, bhh_f, bih_b, bih_f}
  const int GL = 6, GW = 14, JL = 31, JW = 39;  // geek_layer, geek, job_layer, job
  const float* a3W1 = (const float*)d_in[22];
  const float* a3b1 = (const float*)d_in[23];
  const float* a3v  = (const float*)d_in[24];
  const float* a1W1 = (const float*)d_in[25];
  const float* a1b1 = (const float*)d_in[26];
  const float* a1v  = (const float*)d_in[27];
  const float* a2W1 = (const float*)d_in[28];
  const float* a2b1 = (const float*)d_in[29];
  const float* a2v  = (const float*)d_in[30];
  const float* mW1 = (const float*)d_in[47];
  const float* mW2 = (const float*)d_in[48];
  const float* mb1 = (const float*)d_in[49];
  const float* mb2 = (const float*)d_in[50];
  float* out = (float*)d_out;

  // ---- workspace carving ----
  char* p = (char*)d_ws;
  auto alloc = [&](size_t bytes) -> void* {
    void* r = (void*)p;
    p += (bytes + 255) & ~(size_t)255;
    return r;
  };
  _Float16* gv16 = (_Float16*)alloc((size_t)NG * 128 * 2);
  _Float16* jv16 = (_Float16*)alloc((size_t)NG * 128 * 2);
  _Float16* WtGf = (_Float16*)alloc(512 * 256 * 2);
  _Float16* WtGb = (_Float16*)alloc(512 * 256 * 2);
  _Float16* WtJf = (_Float16*)alloc(512 * 256 * 2);
  _Float16* WtJb = (_Float16*)alloc(512 * 256 * 2);
  _Float16* WtJLf = (_Float16*)alloc(512 * 384 * 2);
  _Float16* WtJLb = (_Float16*)alloc(512 * 384 * 2);
  _Float16* WtGLf = (_Float16*)alloc(512 * 384 * 2);
  _Float16* WtGLb = (_Float16*)alloc(512 * 384 * 2);
  _Float16* A1t = (_Float16*)alloc(256 * 256 * 2);
  _Float16* A2t = (_Float16*)alloc(256 * 256 * 2);
  _Float16* A3t = (_Float16*)alloc(256 * 256 * 2);
  _Float16* coWt = (_Float16*)alloc(256 * 256 * 2);
  _Float16* coUt = (_Float16*)alloc(256 * 256 * 2);
  _Float16* W1t768 = (_Float16*)alloc(768 * 768 * 2);
  float* gv_out = (float*)alloc((size_t)NG * 256 * 4);
  float* jv_out = (float*)alloc((size_t)NG * 256 * 4);
  _Float16* cvtbuf = (_Float16*)alloc((size_t)NG * 256 * 2);  // jvout16 then gvout16
  float* tmpBig = (float*)alloc((size_t)NG * 256 * 4);        // attn tmps + Wx (time-disjoint)
  float* Us = (float*)alloc(384 * 256 * 4);
  float* s1 = (float*)alloc(NG * 4);
  float* s2 = (float*)alloc(384 * 4);
  float* s3 = (float*)alloc(384 * 4);
  float* sj = (float*)alloc(384 * 256 * 4);
  _Float16* sj16 = (_Float16*)alloc(384 * 256 * 2);
  float* cj = (float*)alloc(384 * 256 * 4);
  _Float16* cj16 = (_Float16*)alloc(384 * 256 * 2);
  float* sr = (float*)alloc(384 * 256 * 4);
  _Float16* sr16 = (_Float16*)alloc(384 * 256 * 2);
  float* cg = (float*)alloc(384 * 256 * 4);
  _Float16* cg16 = (_Float16*)alloc(384 * 256 * 2);
  float* gj = (float*)alloc(32 * 256 * 4);
  float* gr = (float*)alloc(32 * 256 * 4);
  _Float16* x16 = (_Float16*)alloc(32 * 768 * 2);
  float* hmlp = (float*)alloc(32 * 768 * 4);

  auto wT = [&](const void* src, _Float16* dst, int K, int N, int Ktot,
                int kOff) {
    int total = K * N;
    k_wT16<<<(total + 255) / 256, 256, 0, stream>>>((const float*)src, dst, K,
                                                    N, Ktot, kOff);
  };

  // ---- phase 0: gathers + weight conversion (f16, transposed [N][K]) ----
  k_gather_emb<<<(NG * 128 + 255) / 256, 256, 0, stream>>>(geek_sent, emb, gv16,
                                                           NG * 128);
  k_gather_emb<<<(NG * 128 + 255) / 256, 256, 0, stream>>>(job_sent, emb, jv16,
                                                           NG * 128);
  // word lstms: Wt[n][0..Kx)=Wih, [Kx..Kx+128)=Whh, Kx=128
  wT(d_in[GW + 3], WtGf, 128, 512, 256, 0);   // geek Wih_f
  wT(d_in[GW + 1], WtGf, 128, 512, 256, 128); // geek Whh_f
  wT(d_in[GW + 2], WtGb, 128, 512, 256, 0);
  wT(d_in[GW + 0], WtGb, 128, 512, 256, 128);
  wT(d_in[JW + 3], WtJf, 128, 512, 256, 0);
  wT(d_in[JW + 1], WtJf, 128, 512, 256, 128);
  wT(d_in[JW + 2], WtJb, 128, 512, 256, 0);
  wT(d_in[JW + 0], WtJb, 128, 512, 256, 128);
  // layer lstms: Kx=256
  wT(d_in[JL + 3], WtJLf, 256, 512, 384, 0);
  wT(d_in[JL + 1], WtJLf, 128, 512, 384, 256);
  wT(d_in[JL + 2], WtJLb, 256, 512, 384, 0);
  wT(d_in[JL + 0], WtJLb, 128, 512, 384, 256);
  wT(d_in[GL + 3], WtGLf, 256, 512, 384, 0);
  wT(d_in[GL + 1], WtGLf, 128, 512, 384, 256);
  wT(d_in[GL + 2], WtGLb, 256, 512, 384, 0);
  wT(d_in[GL + 0], WtGLb, 128, 512, 384, 256);
  wT(a1W1, A1t, 256, 256, 256, 0);
  wT(a2W1, A2t, 256, 256, 256, 0);
  wT(a3W1, A3t, 256, 256, 256, 0);
  wT(coW, coWt, 256, 256, 256, 0);
  wT(coU, coUt, 256, 256, 256, 0);
  wT(mW1, W1t768, 768, 768, 768, 0);

  // ---- phase 1: word-level BiLSTMs (fused x@Wih + h@Whh per step) ----
  k_lstm_scan<<<dim3(384 / 16, 2), 128, 0, stream>>>(
      gv16, 128, 50, WtGf, WtGb, (const float*)d_in[GW + 7],
      (const float*)d_in[GW + 5], (const float*)d_in[GW + 6],
      (const float*)d_in[GW + 4], gv_out);
  k_lstm_scan<<<dim3(384 / 16, 2), 128, 0, stream>>>(
      jv16, 128, 50, WtJf, WtJb, (const float*)d_in[JW + 7],
      (const float*)d_in[JW + 5], (const float*)d_in[JW + 6],
      (const float*)d_in[JW + 4], jv_out);

  // ---- phase 2: job attn1 -> sj ----
  k_cvt16<<<(NG * 256 + 255) / 256, 256, 0, stream>>>(jv_out, cvtbuf, NG * 256);
  k_gemm<<<dim3(NG / 64, 4), 128, 0, stream>>>(cvtbuf, A1t, nullptr, tmpBig,
                                               NG, 256, 256, 0);
  k_attn_score<<<NG, 256, 0, stream>>>(tmpBig, a1b1, a1v, s1, NG);
  k_attn_pool<<<384, 256, 0, stream>>>(s1, jv_out, sj, 50);
  k_cvt16<<<(384 * 256 + 255) / 256, 256, 0, stream>>>(sj, sj16, 384 * 256);

  // ---- phase 3: job layer BiLSTM -> cj; attn2 -> gj ----
  k_lstm_scan<<<dim3(2, 2), 128, 0, stream>>>(
      sj16, 256, 12, WtJLf, WtJLb, (const float*)d_in[JL + 7],
      (const float*)d_in[JL + 5], (const float*)d_in[JL + 6],
      (const float*)d_in[JL + 4], cj);
  k_cvt16<<<(384 * 256 + 255) / 256, 256, 0, stream>>>(cj, cj16, 384 * 256);
  k_gemm<<<dim3(384 / 64, 4), 128, 0, stream>>>(cj16, A2t, nullptr, tmpBig,
                                                384, 256, 256, 0);
  k_attn_score<<<384, 256, 0, stream>>>(tmpBig, a2b1, a2v, s2, 384);
  k_attn_pool<<<32, 256, 0, stream>>>(s2, cj, gj, 12);

  // ---- phase 4: co-attention -> sr ----
  k_cvt16<<<(NG * 256 + 255) / 256, 256, 0, stream>>>(gv_out, cvtbuf, NG * 256);
  k_gemm<<<dim3(NG / 64, 4), 128, 0, stream>>>(cvtbuf, coWt, nullptr, tmpBig,
                                               NG, 256, 256, 0);  // Wx
  k_gemm<<<dim3(384 / 64, 4), 128, 0, stream>>>(sj16, coUt, nullptr, Us, 384,
                                                256, 256, 0);
  k_co_attn<<<384, 256, 0, stream>>>(tmpBig, Us, coV, gv_out, sr);
  k_cvt16<<<(384 * 256 + 255) / 256, 256, 0, stream>>>(sr, sr16, 384 * 256);

  // ---- phase 5: geek layer BiLSTM -> cg; self attn -> gr ----
  k_lstm_scan<<<dim3(2, 2), 128, 0, stream>>>(
      sr16, 256, 12, WtGLf, WtGLb, (const float*)d_in[GL + 7],
      (const float*)d_in[GL + 5], (const float*)d_in[GL + 6],
      (const float*)d_in[GL + 4], cg);
  k_cvt16<<<(384 * 256 + 255) / 256, 256, 0, stream>>>(cg, cg16, 384 * 256);
  k_gemm<<<dim3(384 / 64, 4), 128, 0, stream>>>(cg16, A3t, nullptr, tmpBig,
                                                384, 256, 256, 0);
  k_attn_score<<<384, 256, 0, stream>>>(tmpBig, a3b1, a3v, s3, 384);
  k_attn_pool<<<32, 256, 0, stream>>>(s3, cg, gr, 12);

  // ---- phase 6: MLP head ----
  k_concat<<<(32 * 768 + 255) / 256, 256, 0, stream>>>(gj, gr, x16);
  k_gemm<<<dim3(1, 12), 128, 0, stream>>>(x16, W1t768, mb1, hmlp, 32, 768, 768,
                                          1);
  k_final<<<32, 256, 0, stream>>>(hmlp, mW2, mb2, out);
}
